// PointTransformerLayer_55009941127307
// MI455X (gfx1250) — matrix operand order repaired and checked by reading, not compiled, not run on hardware
//
#include <hip/hip_runtime.h>
#include <hip/hip_bf16.h>

// ---------------- problem constants ----------------
#define BATCH 2
#define NPTS  768
#define DIM   64
#define DHID  256          // D*H
#define PHID  64
#define KNN   20
#define BN    (BATCH*NPTS)          // 1536 query points
#define NEDGE (BN*KNN)              // 30720 gathered edges (divisible by 16)
#define MTILES (NEDGE/16)           // 1920

typedef __attribute__((ext_vector_type(16))) _Float16 v16h;
typedef __attribute__((ext_vector_type(8)))  float    v8f;

// ---------------- WMMA fragment helpers ----------------
// A fragment: 16x32 f16 tile, row-major source with row stride ld.
// ISA layout: lanes 0-15 -> M=lane, K base 0 ; lanes 16-31 -> M=lane-16, K base 8
//   vgpr r<4  : K = kb + 2r, kb + 2r + 1
//   vgpr r>=4 : K = 16 + kb + 2(r-4), +1
__device__ __forceinline__ v16h load_a_frag(const _Float16* A, int ld, int lane) {
  const int m  = lane & 15;
  const int kb = (lane & 16) ? 8 : 0;
  const _Float16* row = A + m * ld;
  v16h a;
#pragma unroll
  for (int r = 0; r < 8; ++r) {
    const int k0 = (r < 4) ? (kb + 2 * r) : (16 + kb + 2 * (r - 4));
    a[2 * r]     = row[k0];
    a[2 * r + 1] = row[k0 + 1];
  }
  return a;
}

// B operand is pre-packed so each lane reads 16 contiguous halves (32B load).
// Packed layout per 32x16 k-tile: [lane][16 halves], lane<16 covers K 0..7,16..23 of
// column (lane&15), lane>=16 covers K 8..15,24..31.
__device__ __forceinline__ v16h load_b_frag(const _Float16* Bpacked, int lane) {
  return *(const v16h*)(Bpacked + lane * 16);
}

// C/D layout: vgpr r, lanes 0-15 -> (M=r,   N=lane)
//                     lanes 16-31-> (M=r+8, N=lane-16)
__device__ __forceinline__ void cd_coords(int lane, int r, int& m, int& n) {
  m = r + ((lane & 16) ? 8 : 0);
  n = lane & 15;
}

__device__ __forceinline__ v8f wmma_f16(v16h a, v16h b, v8f c) {
  return __builtin_amdgcn_wmma_f32_16x16x32_f16(false, a, false, b,
                                                (short)0, c, false, false);
}

// ---------------- K0: pack weights into B-fragment layout (f32 -> f16) -------
__device__ __forceinline__ void pack_one(const float* W, _Float16* out,
                                         int local, int ktiles, int ndim) {
  const int tile  = local >> 9;          // /512
  const int lane  = (local >> 4) & 31;
  const int e     = local & 15;
  const int ntile = tile / ktiles;
  const int ktile = tile - ntile * ktiles;
  const int n = ntile * 16 + (lane & 15);
  const int k = ktile * 32 + ((lane & 16) ? 8 : 0) + (e & 7) + ((e & 8) ? 16 : 0);
  out[local] = (_Float16)W[k * ndim + n];
}

__global__ void pack_weights_kernel(const float* Wp2, const float* Wa1, const float* Wa2,
                                    _Float16* p_p2, _Float16* p_a1, _Float16* p_a2) {
  int t = blockIdx.x * blockDim.x + threadIdx.x;
  // Wp2: 64x64  -> (kt=2, nt=4)  -> 8 tiles  = 4096 halves
  // Wa1: 64x256 -> (kt=2, nt=16) -> 32 tiles = 16384 halves
  // Wa2: 256x64 -> (kt=8, nt=4)  -> 32 tiles = 16384 halves
  if (t < 4096)               pack_one(Wp2, p_p2, t,          2, 64);
  else if (t < 4096 + 16384)  pack_one(Wa1, p_a1, t - 4096,   2, 256);
  else if (t < 4096 + 32768)  pack_one(Wa2, p_a2, t - 20480,  8, 64);
}

// ---------------- K1: qkv projection  x[BN,3] @ Wqkv[3,192] -------------------
__global__ void qkv_kernel(const float* x, const float* Wqkv,
                           float* q, float* k, float* v) {
  int t = blockIdx.x * blockDim.x + threadIdx.x;
  if (t >= BN * 192) return;
  const int row = t / 192, col = t - row * 192;
  const float* xr = x + row * 3;
  float acc = xr[0] * Wqkv[col] + xr[1] * Wqkv[192 + col] + xr[2] * Wqkv[384 + col];
  const int sel = col >> 6, d = col & 63;
  float* dst = (sel == 0) ? q : (sel == 1) ? k : v;
  dst[row * DIM + d] = acc;
}

// ---------------- K2: kNN (20 smallest distances, stable tie-break) ----------
__global__ void knn_kernel(const float* pos, int* idx) {
  int bi = blockIdx.x * blockDim.x + threadIdx.x;
  if (bi >= BN) return;
  const int b = bi / NPTS;
  const float px = pos[bi * 3 + 0], py = pos[bi * 3 + 1], pz = pos[bi * 3 + 2];
  float bd[KNN];
  int   bj[KNN];
#pragma unroll
  for (int s = 0; s < KNN; ++s) { bd[s] = 1e30f; bj[s] = 0; }
  const float* pb = pos + (size_t)b * NPTS * 3;
  for (int j = 0; j < NPTS; ++j) {
    const float dx = px - pb[j * 3 + 0];
    const float dy = py - pb[j * 3 + 1];
    const float dz = pz - pb[j * 3 + 2];
    const float d2 = dx * dx + dy * dy + dz * dz;
    if (d2 < bd[KNN - 1]) {          // strict: equal dist keeps earlier index
      bd[KNN - 1] = d2; bj[KNN - 1] = j;
#pragma unroll
      for (int s = KNN - 1; s > 0; --s) {
        if (bd[s] < bd[s - 1]) {     // strict: stable w.r.t. insertion order
          float td = bd[s]; bd[s] = bd[s - 1]; bd[s - 1] = td;
          int   tj = bj[s]; bj[s] = bj[s - 1]; bj[s - 1] = tj;
        }
      }
    }
  }
#pragma unroll
  for (int s = 0; s < KNN; ++s) idx[bi * KNN + s] = bj[s];
}

// ---------------- K3: edge hidden layer h = relu(rel_pos @ Wp1 + b_p1) -------
__global__ void edge_h_kernel(const float* pos, const int* idx,
                              const float* Wp1, const float* bp1, _Float16* h16) {
  int t = blockIdx.x * blockDim.x + threadIdx.x;
  if (t >= NEDGE * PHID) return;
  const int e = t >> 6, d = t & 63;
  const int bi = e / KNN;
  const int b  = bi / NPTS;
  const int j  = idx[e];
  const int jb = b * NPTS + j;
  const float rx = pos[bi * 3 + 0] - pos[jb * 3 + 0];
  const float ry = pos[bi * 3 + 1] - pos[jb * 3 + 1];
  const float rz = pos[bi * 3 + 2] - pos[jb * 3 + 2];
  float hv = rx * Wp1[d] + ry * Wp1[64 + d] + rz * Wp1[128 + d] + bp1[d];
  hv = fmaxf(hv, 0.0f);
  h16[(size_t)e * PHID + d] = (_Float16)hv;
}

// ---------------- K4: pe = h @ Wp2 + b_p2 ; E = q - k + pe (f16) --------------
// One wave per M-tile: A fragments (16x64) cached in VGPRs, looped over 4 ntiles.
__global__ void gemm_pe_kernel(const _Float16* h16, const _Float16* p_p2,
                               const float* bp2, const float* q, const float* k,
                               const int* idx, float* pe_out, _Float16* E16) {
  const int lane  = threadIdx.x & 31;
  const int mtile = blockIdx.x * (blockDim.x >> 5) + (threadIdx.x >> 5);
  if (mtile >= MTILES) return;                   // uniform per wave
  const _Float16* Arow = h16 + (size_t)mtile * 16 * PHID;
  v16h a0 = load_a_frag(Arow, PHID, lane);
  v16h a1 = load_a_frag(Arow + 32, PHID, lane);
#pragma unroll
  for (int ntile = 0; ntile < 4; ++ntile) {
    const _Float16* Bp = p_p2 + (size_t)ntile * 2 * 512;
    v8f acc = {};
    acc = wmma_f16(a0, load_b_frag(Bp, lane), acc);
    acc = wmma_f16(a1, load_b_frag(Bp + 512, lane), acc);
#pragma unroll
    for (int r = 0; r < 8; ++r) {
      int m, n; cd_coords(lane, r, m, n);
      const int e = mtile * 16 + m;
      const int d = ntile * 16 + n;
      const float pe = acc[r] + bp2[d];
      pe_out[(size_t)e * DIM + d] = pe;
      const int bi = e / KNN;
      const int kr = (bi / NPTS) * NPTS + idx[e];
      E16[(size_t)e * DIM + d] = (_Float16)(q[bi * DIM + d] - k[kr * DIM + d] + pe);
    }
  }
}

// ---------------- K5: S1 = relu(E @ Wa1 + b_a1) (f16) ------------------------
// One wave per M-tile: A cached once, 16 ntiles x 2 ktiles = 32 WMMAs.
__global__ void gemm_a1_kernel(const _Float16* E16, const _Float16* p_a1,
                               const float* ba1, _Float16* S1) {
  const int lane  = threadIdx.x & 31;
  const int mtile = blockIdx.x * (blockDim.x >> 5) + (threadIdx.x >> 5);
  if (mtile >= MTILES) return;
  const _Float16* Arow = E16 + (size_t)mtile * 16 * DIM;
  v16h a0 = load_a_frag(Arow, DIM, lane);
  v16h a1 = load_a_frag(Arow + 32, DIM, lane);
#pragma unroll
  for (int ntile = 0; ntile < 16; ++ntile) {
    const _Float16* Bp = p_a1 + (size_t)ntile * 2 * 512;
    v8f acc = {};
    acc = wmma_f16(a0, load_b_frag(Bp, lane), acc);
    acc = wmma_f16(a1, load_b_frag(Bp + 512, lane), acc);
#pragma unroll
    for (int r = 0; r < 8; ++r) {
      int m, n; cd_coords(lane, r, m, n);
      const int e = mtile * 16 + m;
      const int d = ntile * 16 + n;
      S1[(size_t)e * DHID + d] = (_Float16)fmaxf(acc[r] + ba1[d], 0.0f);
    }
  }
}

// ---------------- K6: sim = S1 @ Wa2 + b_a2 (f32) ----------------------------
// One wave per M-tile: 8 A fragments (16x256) cached, 4 ntiles x 8 kt = 32 WMMAs.
__global__ void gemm_a2_kernel(const _Float16* S1, const _Float16* p_a2,
                               const float* ba2, float* sim) {
  const int lane  = threadIdx.x & 31;
  const int mtile = blockIdx.x * (blockDim.x >> 5) + (threadIdx.x >> 5);
  if (mtile >= MTILES) return;
  const _Float16* Arow = S1 + (size_t)mtile * 16 * DHID;
  v16h a[8];
#pragma unroll
  for (int kt = 0; kt < 8; ++kt) a[kt] = load_a_frag(Arow + kt * 32, DHID, lane);
#pragma unroll
  for (int ntile = 0; ntile < 4; ++ntile) {
    const _Float16* Bp = p_a2 + (size_t)ntile * 8 * 512;
    v8f acc = {};
#pragma unroll
    for (int kt = 0; kt < 8; ++kt)
      acc = wmma_f16(a[kt], load_b_frag(Bp + kt * 512, lane), acc);
#pragma unroll
    for (int r = 0; r < 8; ++r) {
      int m, n; cd_coords(lane, r, m, n);
      const int e = mtile * 16 + m;
      const int d = ntile * 16 + n;
      sim[(size_t)e * DIM + d] = acc[r] + ba2[d];
    }
  }
}

// ---------------- K7: softmax over K + weighted sum --------------------------
__global__ void softmax_out_kernel(const float* sim, const float* pe,
                                   const float* v, const int* idx, float* out) {
  int t = blockIdx.x * blockDim.x + threadIdx.x;
  if (t >= BN * DIM) return;
  const int bi = t >> 6, d = t & 63;
  const int base = bi * KNN;
  const int bb = (bi / NPTS) * NPTS;
  float mx = -1e30f;
#pragma unroll
  for (int kk = 0; kk < KNN; ++kk)
    mx = fmaxf(mx, sim[(size_t)(base + kk) * DIM + d]);
  float den = 0.0f, num = 0.0f;
#pragma unroll
  for (int kk = 0; kk < KNN; ++kk) {
    const int e = base + kk;
    const float w = __expf(sim[(size_t)e * DIM + d] - mx);
    const float vg = v[(bb + idx[e]) * DIM + d] + pe[(size_t)e * DIM + d];
    den += w;
    num += w * vg;
  }
  out[bi * DIM + d] = num / den;
}

// ---------------- host launch -------------------------------------------------
extern "C" void kernel_launch(void* const* d_in, const int* in_sizes, int n_in,
                              void* d_out, int out_size, void* d_ws, size_t ws_size,
                              hipStream_t stream) {
  const float* x     = (const float*)d_in[0];
  const float* pos   = (const float*)d_in[1];
  const float* W_qkv = (const float*)d_in[2];
  const float* W_p1  = (const float*)d_in[3];
  const float* b_p1  = (const float*)d_in[4];
  const float* W_p2  = (const float*)d_in[5];
  const float* b_p2  = (const float*)d_in[6];
  const float* W_a1  = (const float*)d_in[7];
  const float* b_a1  = (const float*)d_in[8];
  const float* W_a2  = (const float*)d_in[9];
  const float* b_a2  = (const float*)d_in[10];
  float* out = (float*)d_out;

  char* ws = (char*)d_ws;
  size_t cur = 0;
  auto alloc = [&](size_t bytes) {
    char* p = ws + cur;
    cur += (bytes + 255) & ~(size_t)255;
    return p;
  };
  float*    q    = (float*)alloc((size_t)BN * DIM * 4);
  float*    k    = (float*)alloc((size_t)BN * DIM * 4);
  float*    v    = (float*)alloc((size_t)BN * DIM * 4);
  int*      idx  = (int*)alloc((size_t)BN * KNN * 4);
  _Float16* h16  = (_Float16*)alloc((size_t)NEDGE * PHID * 2);
  float*    pe   = (float*)alloc((size_t)NEDGE * DIM * 4);
  _Float16* E16  = (_Float16*)alloc((size_t)NEDGE * DIM * 2);
  _Float16* S1   = (_Float16*)alloc((size_t)NEDGE * DHID * 2);
  float*    sim  = (float*)alloc((size_t)NEDGE * DIM * 4);
  _Float16* p_p2 = (_Float16*)alloc(4096 * 2);
  _Float16* p_a1 = (_Float16*)alloc(16384 * 2);
  _Float16* p_a2 = (_Float16*)alloc(16384 * 2);

  const int T = 256;
  pack_weights_kernel<<<(36864 + T - 1) / T, T, 0, stream>>>(W_p2, W_a1, W_a2,
                                                             p_p2, p_a1, p_a2);
  qkv_kernel<<<(BN * 192 + T - 1) / T, T, 0, stream>>>(x, W_qkv, q, k, v);
  knn_kernel<<<(BN + T - 1) / T, T, 0, stream>>>(pos, idx);
  edge_h_kernel<<<(NEDGE * PHID + T - 1) / T, T, 0, stream>>>(pos, idx, W_p1, b_p1, h16);

  const int WPB = T / 32;  // waves per block; one wave per 16-row M-tile
  gemm_pe_kernel<<<(MTILES + WPB - 1) / WPB, T, 0, stream>>>(h16, p_p2, b_p2,
                                                             q, k, idx, pe, E16);
  gemm_a1_kernel<<<(MTILES + WPB - 1) / WPB, T, 0, stream>>>(E16, p_a1, b_a1, S1);
  gemm_a2_kernel<<<(MTILES + WPB - 1) / WPB, T, 0, stream>>>(S1, p_a2, b_a2, sim);

  softmax_out_kernel<<<(BN * DIM + T - 1) / T, T, 0, stream>>>(sim, pe, v, idx, out);
}
